// PM_MoE_part_lvl_MLP_Block_33612414058826
// MI455X (gfx1250) — matrix operand order repaired and checked by reading, not compiled, not run on hardware
//
#include <hip/hip_runtime.h>
#include <hip/hip_bf16.h>

// ---------------------------------------------------------------------------
// MoE 3-layer expert MLP for MI455X (gfx1250, wave32, bf16 WMMA +
// double-buffered async global->LDS staging).
//   T=32768 tokens, F=256, H=1024, OUT=256, E=8.
//   0) bucket tokens by expert
//   1) one-time f32->bf16: x -> xbf, weights -> transposed [N][K] bf16
//   2) L1: h1 = silu(x@Wg+bg)*(x@Wu+bu)   (M_TILE=64, dual accumulators)
//   3) L2: h2 = relu(h1@W1+b1)            (M_TILE=128)
//   4) L3: out = h2@W2+b2 -> f32 scatter  (M_TILE=128)
// ---------------------------------------------------------------------------

#define T_TOT   32768
#define F_DIM   256
#define H_DIM   1024
#define OUT_DIM 256
#define NEXP    8
#define NTHREADS 256

typedef __attribute__((ext_vector_type(16))) __bf16 v16bf;
typedef __attribute__((ext_vector_type(8)))  __bf16 v8bf;
typedef __attribute__((ext_vector_type(8)))  float  v8f;

// LDS tile stride: 40 bf16 = 80B per row/col -> 16B-aligned fragment reads,
// +20 banks per row => conflict-spread across the 64 banks.
#define LDS_STRIDE 40

__device__ __forceinline__ v16bf cat8(v8bf lo, v8bf hi) {
    return __builtin_shufflevector(lo, hi, 0,1,2,3,4,5,6,7,8,9,10,11,12,13,14,15);
}

__device__ __forceinline__ v8f wmma_bf16(v16bf a, v16bf b, v8f c) {
    return __builtin_amdgcn_wmma_f32_16x16x32_bf16(false, a, false, b,
                                                   (short)0, c, false, false);
}

// Async 16B global->LDS copy (ASYNCcnt-tracked, no VGPR round trip).
__device__ __forceinline__ void async_copy16(unsigned int ldsOff,
                                             unsigned long long gaddr) {
    asm volatile("global_load_async_to_lds_b128 %0, %1, off"
                 :: "v"(ldsOff), "v"(gaddr) : "memory");
}
__device__ __forceinline__ void wait_async0() {
    asm volatile("s_wait_asynccnt 0x0" ::: "memory");
}
// Generic-pointer low 32 bits == LDS byte offset (aperture truncation rule).
__device__ __forceinline__ unsigned int lds_off(const void* p) {
    return (unsigned int)(unsigned long long)p;
}

// ---------------------------------------------------------------------------
// Kernel 0: bucket token ids by selected expert.
// ---------------------------------------------------------------------------
__global__ void bucket_kernel(const int* __restrict__ sel,
                              int* __restrict__ counts,
                              int* __restrict__ idx) {
    int t = blockIdx.x * blockDim.x + threadIdx.x;
    if (t < T_TOT) {
        int e = sel[t] & (NEXP - 1);
        int p = atomicAdd(&counts[e], 1);
        idx[e * T_TOT + p] = t;
    }
}

// ---------------------------------------------------------------------------
// Kernel 1a: elementwise f32 -> bf16 (activations).
// ---------------------------------------------------------------------------
__global__ void cvt_bf16_kernel(const float* __restrict__ in,
                                __bf16* __restrict__ out, int n) {
    int i = (blockIdx.x * blockDim.x + threadIdx.x) * 2;
    if (i < n) {
        out[i]     = (__bf16)in[i];
        out[i + 1] = (__bf16)in[i + 1];
    }
}

// ---------------------------------------------------------------------------
// Kernel 1b: weight convert+transpose: [E][K][N] f32 -> [E][N][K] bf16.
// ---------------------------------------------------------------------------
template<int K, int N>
__global__ void cvt_transpose_kernel(const float* __restrict__ in,
                                     __bf16* __restrict__ out) {
    __shared__ float tile[32][33];
    const int e  = blockIdx.z;
    const int kb = blockIdx.y * 32;
    const int nb = blockIdx.x * 32;
    const float* src = in + (size_t)e * K * N;
    __bf16* dst = out + (size_t)e * N * K;
    const int tx = threadIdx.x, ty = threadIdx.y;   // (32, 8)
#pragma unroll
    for (int j = ty; j < 32; j += 8)
        tile[j][tx] = src[(size_t)(kb + j) * N + (nb + tx)];
    __syncthreads();
#pragma unroll
    for (int j = ty; j < 32; j += 8)
        dst[(size_t)(nb + j) * K + (kb + tx)] = (__bf16)tile[tx][j];
}

// ---------------------------------------------------------------------------
// Generic bucketed GEMM. Block = (MSUB*16) tokens x 128 cols, 8 wave32s.
// Each wave owns 16 output columns; B fragment reused across MSUB WMMAs.
// All staging addresses are hoisted out of the K loop; per step a copy is
// just async_copy(ldsOff[buf][j], base[j] + 2*k0).
//   MODE 1: SwiGLU (dual B/bias, silu(g)*u -> bf16)
//   MODE 2: ReLU -> bf16
//   MODE 3: bias -> f32 scatter
// A: [T][KD] bf16 (gathered by idx); BT*: [E][ND][KD] bf16 (pre-transposed).
// KD must be a multiple of 64.
// ---------------------------------------------------------------------------
template<int KD, int ND, int MSUB, int MODE>
__global__ __launch_bounds__(NTHREADS)
void moe_gemm(const __bf16* __restrict__ A,
              const __bf16* __restrict__ BT0,
              const __bf16* __restrict__ BT1,
              const float* __restrict__ bias0,
              const float* __restrict__ bias1,
              const int* __restrict__ counts,
              const int* __restrict__ idx,
              __bf16* __restrict__ outBf,
              float* __restrict__ outF) {
    constexpr int M_TILE = MSUB * 16;
    constexpr int NA = (M_TILE * 4) / NTHREADS;   // A copies per thread (1|2)
    constexpr int NB = (128 * 4) / NTHREADS;      // B copies per thread (2)

    const int e   = blockIdx.z;
    const int cnt = counts[e];
    const int m0  = blockIdx.y * M_TILE;
    if (m0 >= cnt) return;                 // block-uniform: EXEC all-1s below
    const int nb  = blockIdx.x * 128;

    __shared__ __bf16 ldsA [2][M_TILE * LDS_STRIDE];
    __shared__ __bf16 ldsB0[2][128 * LDS_STRIDE];
    __shared__ __bf16 ldsB1[2][(MODE == 1 ? 128 : 8) * LDS_STRIDE];
    __shared__ int    ldsRow[M_TILE];

    const int tid = threadIdx.x;
    if (tid < M_TILE) {
        int m = m0 + tid;
        ldsRow[tid] = (m < cnt) ? idx[e * T_TOT + m] : -1;
    }
    __syncthreads();                        // ldsRow visible to all

    const int lane  = tid & 31;
    const int wave  = tid >> 5;
    const int lm    = lane & 15;
    const int hiH   = lane >> 4;
    const int khalf = hiH << 3;             // unused (A frag uses kbase)
    const int kbase = hiH << 4;             // A/B frag K offset: 0 / 16
    const int col   = wave * 16 + lm;

    const __bf16* Bt0 = BT0 + (size_t)e * ND * KD;
    const __bf16* Bt1 = BT1 + (size_t)e * ND * KD;

    // ---- Hoisted staging addresses (k-invariant). Padded rows clamp to row
    // 0 (valid address; their outputs are masked at the epilogue). ----
    unsigned long long gaA[NA], gaB0[NB], gaB1[NB];
    unsigned int lA[2][NA], lB0[2][NB], lB1[2][NB];
#pragma unroll
    for (int j = 0; j < NA; ++j) {
        int i = tid + j * NTHREADS;
        int m = i >> 2, kq = (i & 3) * 8;
        int row = ldsRow[m]; row = row < 0 ? 0 : row;
        gaA[j] = (unsigned long long)(const void*)(A + (size_t)row * KD + kq);
        lA[0][j] = lds_off(&ldsA[0][m * LDS_STRIDE + kq]);
        lA[1][j] = lds_off(&ldsA[1][m * LDS_STRIDE + kq]);
    }
#pragma unroll
    for (int j = 0; j < NB; ++j) {
        int i = tid + j * NTHREADS;
        int n = i >> 2, kq = (i & 3) * 8;
        gaB0[j] = (unsigned long long)(const void*)(Bt0 + (size_t)(nb + n) * KD + kq);
        lB0[0][j] = lds_off(&ldsB0[0][n * LDS_STRIDE + kq]);
        lB0[1][j] = lds_off(&ldsB0[1][n * LDS_STRIDE + kq]);
        if constexpr (MODE == 1) {
            gaB1[j] = (unsigned long long)(const void*)(Bt1 + (size_t)(nb + n) * KD + kq);
            lB1[0][j] = lds_off(&ldsB1[0][n * LDS_STRIDE + kq]);
            lB1[1][j] = lds_off(&ldsB1[1][n * LDS_STRIDE + kq]);
        }
    }

    v8f acc0[MSUB]; v8f acc1[MSUB];
#pragma unroll
    for (int t = 0; t < MSUB; ++t) { acc0[t] = {}; acc1[t] = {}; }

    auto stageStep = [&](int k0, int buf) {
        const unsigned long long ko = (unsigned long long)(k0 * 2);  // bytes
#pragma unroll
        for (int j = 0; j < NA; ++j)
            async_copy16(lA[buf][j], gaA[j] + ko);
#pragma unroll
        for (int j = 0; j < NB; ++j) {
            async_copy16(lB0[buf][j], gaB0[j] + ko);
            if constexpr (MODE == 1)
                async_copy16(lB1[buf][j], gaB1[j] + ko);
        }
    };

    auto computeStep = [&](int buf) {
        const __bf16* pb0 = &ldsB0[buf][col * LDS_STRIDE + kbase];
        v16bf b0 = cat8(*(const v8bf*)pb0, *(const v8bf*)(pb0 + 8));
        v16bf b1;
        if constexpr (MODE == 1) {
            const __bf16* pb1 = &ldsB1[buf][col * LDS_STRIDE + kbase];
            b1 = cat8(*(const v8bf*)pb1, *(const v8bf*)(pb1 + 8));
        }
#pragma unroll
        for (int t = 0; t < MSUB; ++t) {
            const __bf16* pa = &ldsA[buf][(t * 16 + lm) * LDS_STRIDE + kbase];
            v16bf a = cat8(*(const v8bf*)pa, *(const v8bf*)(pa + 8));
            acc0[t] = wmma_bf16(a, b0, acc0[t]);
            if constexpr (MODE == 1)
                acc1[t] = wmma_bf16(a, b1, acc1[t]);
        }
    };

    // Double-buffered pipeline: stage k+32 while computing k.
    stageStep(0, 0);
    for (int k0 = 0; k0 < KD; k0 += 64) {
        wait_async0();
        __syncthreads();
        if (k0 + 32 < KD) stageStep(k0 + 32, 1);
        computeStep(0);

        wait_async0();
        __syncthreads();
        if (k0 + 64 < KD) stageStep(k0 + 64, 0);
        computeStep(1);
    }

    // Epilogue. C layout: VGPR r -> M = r + 8*hiH, N = lane&15.
    const int nGlob = nb + col;
    const float bv0 = bias0[e * ND + nGlob];
    float bv1 = 0.0f;
    if constexpr (MODE == 1) bv1 = bias1[e * ND + nGlob];
    const int mofs = hiH << 3;
#pragma unroll
    for (int t = 0; t < MSUB; ++t) {
#pragma unroll
        for (int r = 0; r < 8; ++r) {
            int row = ldsRow[t * 16 + r + mofs];
            if (row < 0) continue;
            float v = acc0[t][r] + bv0;
            if constexpr (MODE == 1) {
                float u = acc1[t][r] + bv1;
                float s = v * (1.0f / (1.0f + __expf(-v)));   // silu
                outBf[(size_t)row * ND + nGlob] = (__bf16)(s * u);
            } else if constexpr (MODE == 2) {
                v = v > 0.0f ? v : 0.0f;                      // relu
                outBf[(size_t)row * ND + nGlob] = (__bf16)v;
            } else {
                outF[(size_t)row * ND + nGlob] = v;
            }
        }
    }
}

// ---------------------------------------------------------------------------
// Host launcher.
// Workspace: counts(256B) | idx(1MB) | xbf(16MB) | WgT(4MB) | WuT(4MB) |
//            W1T(16MB) | W2T(4MB) | h1(64MB) | h2(64MB)  ~= 173MB.
// ---------------------------------------------------------------------------
extern "C" void kernel_launch(void* const* d_in, const int* in_sizes, int n_in,
                              void* d_out, int out_size, void* d_ws, size_t ws_size,
                              hipStream_t stream) {
    const float* x   = (const float*)d_in[0];
    const int*   sel = (const int*)d_in[1];   // JAX x64-off => int32
    const float* Wg  = (const float*)d_in[2];
    const float* bg  = (const float*)d_in[3];
    const float* Wu  = (const float*)d_in[4];
    const float* bu  = (const float*)d_in[5];
    const float* W1  = (const float*)d_in[6];
    const float* b1  = (const float*)d_in[7];
    const float* W2  = (const float*)d_in[8];
    const float* b2  = (const float*)d_in[9];
    float* out = (float*)d_out;

    char* p = (char*)d_ws;
    int* counts = (int*)p;                       p += 256;
    int* idx    = (int*)p;                       p += (size_t)NEXP * T_TOT * 4;
    __bf16* xbf = (__bf16*)p;                    p += (size_t)T_TOT * F_DIM * 2;
    __bf16* WgT = (__bf16*)p;                    p += (size_t)NEXP * F_DIM * H_DIM * 2;
    __bf16* WuT = (__bf16*)p;                    p += (size_t)NEXP * F_DIM * H_DIM * 2;
    __bf16* W1T = (__bf16*)p;                    p += (size_t)NEXP * H_DIM * H_DIM * 2;
    __bf16* W2T = (__bf16*)p;                    p += (size_t)NEXP * H_DIM * OUT_DIM * 2;
    __bf16* h1  = (__bf16*)p;                    p += (size_t)T_TOT * H_DIM * 2;
    __bf16* h2  = (__bf16*)p;

    hipMemsetAsync(counts, 0, NEXP * sizeof(int), stream);
    bucket_kernel<<<dim3(T_TOT / 256), dim3(256), 0, stream>>>(sel, counts, idx);

    {
        int n = T_TOT * F_DIM;
        cvt_bf16_kernel<<<dim3(n / 512), dim3(256), 0, stream>>>(x, xbf, n);
    }
    dim3 tb(32, 8);
    cvt_transpose_kernel<F_DIM, H_DIM>
        <<<dim3(H_DIM / 32, F_DIM / 32, NEXP), tb, 0, stream>>>(Wg, WgT);
    cvt_transpose_kernel<F_DIM, H_DIM>
        <<<dim3(H_DIM / 32, F_DIM / 32, NEXP), tb, 0, stream>>>(Wu, WuT);
    cvt_transpose_kernel<H_DIM, H_DIM>
        <<<dim3(H_DIM / 32, H_DIM / 32, NEXP), tb, 0, stream>>>(W1, W1T);
    cvt_transpose_kernel<H_DIM, OUT_DIM>
        <<<dim3(OUT_DIM / 32, H_DIM / 32, NEXP), tb, 0, stream>>>(W2, W2T);

    dim3 blk(NTHREADS);
    moe_gemm<F_DIM, H_DIM, 4, 1>
        <<<dim3(H_DIM / 128, T_TOT / 64, NEXP), blk, 0, stream>>>(
            xbf, WgT, WuT, bg, bu, counts, idx, h1, nullptr);
    moe_gemm<H_DIM, H_DIM, 8, 2>
        <<<dim3(H_DIM / 128, T_TOT / 128, NEXP), blk, 0, stream>>>(
            h1, W1T, W1T, b1, b1, counts, idx, h2, nullptr);
    moe_gemm<H_DIM, OUT_DIM, 8, 3>
        <<<dim3(OUT_DIM / 128, T_TOT / 128, NEXP), blk, 0, stream>>>(
            h2, W2T, W2T, b2, b2, counts, idx, nullptr, out);
}
